// Encoder_69939247448302
// MI455X (gfx1250) — matrix operand order repaired and checked by reading, not compiled, not run on hardware
//
#include <hip/hip_runtime.h>
#include <hip/hip_bf16.h>
#include <math.h>

typedef __attribute__((ext_vector_type(16))) _Float16 v16h;
typedef __attribute__((ext_vector_type(8)))  _Float16 v8h;
typedef __attribute__((ext_vector_type(8)))  float    v8f;

#define WMMA_F16(A, B, C) \
  __builtin_amdgcn_wmma_f32_16x16x32_f16(false, (A), false, (B), (short)0, (C), false, false)

// ---------------- problem constants ----------------
#define NB   64
#define TIN  2048
#define CIN0 9
#define T1   2025   // 2048-24+1
#define T2   2010   // 2025-16+1
#define T3   2003   // 2010-8+1
#define M1   (NB*T1)   // 129600
#define M2   (NB*T2)   // 128640
#define M3   (NB*T3)   // 128192

// ---- WMMA fragment loaders (16x16x32 f16; ISA §7.12.2 layouts) ------------
// A lane (m = lane&15, hl = lane>>4): slots 0..7 -> K = 8*hl + 0..7,
// slots 8..15 -> K = 16 + 8*hl + 0..7. Both 16B-contiguous chunks.
__device__ __forceinline__ v16h frag_A(const _Float16* p, int hl) {
  v8h lo = *(const v8h*)(p + hl * 8);
  v8h hi = *(const v8h*)(p + 16 + hl * 8);
  return __builtin_shufflevector(lo, hi, 0, 1, 2, 3, 4, 5, 6, 7,
                                         8, 9, 10, 11, 12, 13, 14, 15);
}
// B lane (n = lane&15, hl): slots 0..7 -> K = 8*hl + 0..7, slots 8..15 -> K = 16 + 8*hl + 0..7.
__device__ __forceinline__ v16h frag_B(const _Float16* p, int hl) {
  v8h lo = *(const v8h*)(p + hl * 8);
  v8h hi = *(const v8h*)(p + 16 + hl * 8);
  return __builtin_shufflevector(lo, hi, 0, 1, 2, 3, 4, 5, 6, 7,
                                         8, 9, 10, 11, 12, 13, 14, 15);
}

// ---------------- pooled init (relu >= 0 -> 0.0f identity for max) ---------
__global__ void init_pooled(float* pooled) {
  int i = blockIdx.x * blockDim.x + threadIdx.x;
  if (i < NB * 96) pooled[i] = 0.0f;
}

// -------- weight repack: conv w[o][c][kk] -> f16 wpk[o][k = kk*CIN + c] ----
template <int CINW, int KW, int SH>
__global__ void repack_conv_w(const float* __restrict__ w,
                              _Float16* __restrict__ wpk, int Cout) {
  const int K = CINW * KW;
  int i = blockIdx.x * blockDim.x + threadIdx.x;
  if (i >= Cout * K) return;
  int o = i / K, k = i - o * K;
  int c = k & (CINW - 1), kk = k >> SH;
  wpk[i] = (_Float16)w[(o * CINW + c) * KW + kk];
}

// -------- weight repack: dense f32 -> f16 (layout preserved, K-major) ------
__global__ void cvt_w_f16(const float* __restrict__ w,
                          _Float16* __restrict__ wpk, int n) {
  int i = blockIdx.x * blockDim.x + threadIdx.x;
  if (i < n) wpk[i] = (_Float16)w[i];
}

// ---------------- conv1: direct VALU (K=216, 3.5% of FLOPs) ----------------
__global__ void conv1_direct(const float* __restrict__ x,
                             const float* __restrict__ w1,
                             const float* __restrict__ b1,
                             _Float16* __restrict__ h1) {
  int r = blockIdx.x * blockDim.x + threadIdx.x;
  if (r >= M1) return;
  int b = r / T1, t = r - b * T1;
  const float* xp = x + ((long)b * TIN + t) * CIN0;
  float acc[32];
#pragma unroll
  for (int o = 0; o < 32; ++o) acc[o] = b1[o];
  for (int c = 0; c < CIN0; ++c) {
    float xv[24];
#pragma unroll
    for (int kk = 0; kk < 24; ++kk) xv[kk] = xp[kk * CIN0 + c];
    for (int o = 0; o < 32; ++o) {
      const float* wp = w1 + (o * CIN0 + c) * 24;
      float s = 0.f;
#pragma unroll
      for (int kk = 0; kk < 24; ++kk) s = fmaf(xv[kk], wp[kk], s);
      acc[o] += s;
    }
  }
  _Float16* out = h1 + (long)r * 32;
#pragma unroll
  for (int o = 0; o < 32; ++o) {
    float v = acc[o];
    out[o] = (_Float16)(v > 0.f ? v : 0.f);
  }
}

// ------------- conv2/conv3 as implicit GEMM with WMMA ----------------------
// k = kk*CIN + c matches the [time][CIN] activation layout, so the A operand
// is CONTIGUOUS: addr = inbase + k. Weights are pre-swizzled to the same k
// ordering in f16, so B is contiguous too. One wave owns a 16-row stripe and
// all NT column tiles; A fragments are loaded once per K-step, reused NT x.
template <int CINW, int KW, int NT>
__global__ void conv_gemm(const _Float16* __restrict__ in,
                          const _Float16* __restrict__ wpk,
                          const float* __restrict__ bias,
                          _Float16* __restrict__ out,
                          int Tin, int Tout, int Cout) {
  const int lane = threadIdx.x;
  const int m = lane & 15, hl = lane >> 4;
  const int row = blockIdx.x * 16 + m;
  const int b = row / Tout, t = row - b * Tout;
  const _Float16* abase = in + ((long)b * Tin + t) * CINW;
  constexpr int K = CINW * KW;

  v8f acc[NT] = {};
  for (int k0 = 0; k0 < K; k0 += 32) {
    v16h Af = frag_A(abase + k0, hl);
#pragma unroll
    for (int nt = 0; nt < NT; ++nt) {
      const _Float16* wb = wpk + (long)(nt * 16 + m) * K + k0;
      acc[nt] = WMMA_F16(Af, frag_B(wb, hl), acc[nt]);
    }
  }
#pragma unroll
  for (int nt = 0; nt < NT; ++nt) {
    int col = nt * 16 + m;
    float bv = bias[col];
#pragma unroll
    for (int j = 0; j < 8; ++j) {
      int r = blockIdx.x * 16 + j + 8 * hl;
      float v = acc[nt][j] + bv;
      out[(long)r * Cout + col] = (_Float16)(v > 0.f ? v : 0.f);
    }
  }
}

// ------------- qe: GEMM + bias + relu, M3 x 128, K=96 ----------------------
__global__ void qe_gemm(const _Float16* __restrict__ h3,
                        const _Float16* __restrict__ qw,
                        const float* __restrict__ qb,
                        _Float16* __restrict__ q) {
  const int lane = threadIdx.x;
  const int m = lane & 15, hl = lane >> 4;
  const _Float16* abase = h3 + (long)(blockIdx.x * 16 + m) * 96;

  v8f acc[8] = {};
  for (int k0 = 0; k0 < 96; k0 += 32) {
    v16h Af = frag_A(abase + k0, hl);
#pragma unroll
    for (int nt = 0; nt < 8; ++nt) {
      const _Float16* wb = qw + (long)(nt * 16 + m) * 96 + k0;
      acc[nt] = WMMA_F16(Af, frag_B(wb, hl), acc[nt]);
    }
  }
#pragma unroll
  for (int nt = 0; nt < 8; ++nt) {
    int col = nt * 16 + m;
    float bv = qb[col];
#pragma unroll
    for (int j = 0; j < 8; ++j) {
      int r = blockIdx.x * 16 + j + 8 * hl;
      float v = acc[nt][j] + bv;
      q[(long)r * 128 + col] = (_Float16)(v > 0.f ? v : 0.f);
    }
  }
}

// ------- wp GEMM (K=128, N=640) fused with per-group argmax + codebook -----
// 40 column tiles is too many live accumulators, so instead cache all four
// A fragments in registers and stream the column tiles (q traffic / 40).
__global__ void wp_argmax_quant(const _Float16* __restrict__ q,
                                const _Float16* __restrict__ wpk,
                                const float* __restrict__ wpb,
                                const float* __restrict__ codebook,
                                float* __restrict__ quant,
                                _Float16* __restrict__ quantH,
                                float* __restrict__ targets) {
  __shared__ float tile[16][17];
  const int lane = threadIdx.x;
  const int m = lane & 15, hl = lane >> 4;
  const int mtile = blockIdx.x;
  const _Float16* abase = q + (long)(mtile * 16 + m) * 128;

  v16h Ac[4];
#pragma unroll
  for (int ks = 0; ks < 4; ++ks) Ac[ks] = frag_A(abase + ks * 32, hl);

  float mv0 = -INFINITY, mv1 = -INFINITY;
  int mi0 = 0, mi1 = 0;

  for (int n0 = 0; n0 < 640; n0 += 16) {
    const _Float16* wb = wpk + (long)(n0 + m) * 128;
    v8f acc = {};
#pragma unroll
    for (int ks = 0; ks < 4; ++ks)
      acc = WMMA_F16(Ac[ks], frag_B(wb + ks * 32, hl), acc);
    __syncthreads();                 // previous tile fully consumed
#pragma unroll
    for (int j = 0; j < 8; ++j) tile[j + 8 * hl][m] = acc[j];
    __syncthreads();
    if (lane < 16) {                 // lane owns one row, scans 16 logits
      for (int cc = 0; cc < 16; ++cc) {
        float v = tile[lane][cc] + wpb[n0 + cc];
        if (n0 < 320) {
          if (v > mv0) { mv0 = v; mi0 = n0 + cc; }
        } else {
          if (v > mv1) { mv1 = v; mi1 = n0 + cc - 320; }
        }
      }
    }
  }

  if (lane < 16) {
    long row = (long)mtile * 16 + lane;
    targets[row * 2 + 0] = (float)mi0;
    targets[row * 2 + 1] = (float)mi1;
    float* qr = quant + row * 128;
    _Float16* qh = quantH + row * 128;
    const float* cb0 = codebook + (long)mi0 * 64;           // group 0
    const float* cb1 = codebook + (long)(320 + mi1) * 64;   // group 1
    for (int d = 0; d < 64; ++d) {
      float v0 = cb0[d], v1 = cb1[d];
      qr[d] = v0;       qr[64 + d] = v1;
      qh[d] = (_Float16)v0; qh[64 + d] = (_Float16)v1;
    }
  }
}

// ---------- de GEMM (K=128, N=96) fused with max-pool over time ------------
__global__ void de_pool(const _Float16* __restrict__ quantH,
                        const _Float16* __restrict__ dw,
                        const float* __restrict__ db,
                        float* __restrict__ pooled) {
  const int lane = threadIdx.x;
  const int m = lane & 15, hl = lane >> 4;
  const _Float16* abase = quantH + (long)(blockIdx.x * 16 + m) * 128;

  v8f acc[6] = {};
  for (int k0 = 0; k0 < 128; k0 += 32) {
    v16h Af = frag_A(abase + k0, hl);
#pragma unroll
    for (int nt = 0; nt < 6; ++nt) {
      const _Float16* wb = dw + (long)(nt * 16 + m) * 128 + k0;
      acc[nt] = WMMA_F16(Af, frag_B(wb, hl), acc[nt]);
    }
  }
#pragma unroll
  for (int nt = 0; nt < 6; ++nt) {
    int col = nt * 16 + m;
    float bv = db[col];
#pragma unroll
    for (int j = 0; j < 8; ++j) {
      int r = blockIdx.x * 16 + j + 8 * hl;
      int b = r / T3;
      float v = acc[nt][j] + bv;
      v = v > 0.f ? v : 0.f;
      // v >= 0 and pooled initialized to 0.0f: IEEE bits of non-negative
      // floats order like unsigned ints -> bitwise atomicMax is exact.
      atomicMax((unsigned int*)(pooled + b * 96 + col), __float_as_uint(v));
    }
  }
}

// ---------------------------------------------------------------------------
extern "C" void kernel_launch(void* const* d_in, const int* in_sizes, int n_in,
                              void* d_out, int out_size, void* d_ws, size_t ws_size,
                              hipStream_t stream) {
  (void)in_sizes; (void)n_in; (void)out_size; (void)ws_size;
  const float* x    = (const float*)d_in[0];
  const float* c1w  = (const float*)d_in[1];
  const float* c1b  = (const float*)d_in[2];
  const float* c2w  = (const float*)d_in[3];
  const float* c2b  = (const float*)d_in[4];
  const float* c3w  = (const float*)d_in[5];
  const float* c3b  = (const float*)d_in[6];
  const float* qew  = (const float*)d_in[7];
  const float* qeb  = (const float*)d_in[8];
  const float* wpw  = (const float*)d_in[9];
  const float* wpb  = (const float*)d_in[10];
  const float* cbk  = (const float*)d_in[11];
  const float* dew  = (const float*)d_in[12];
  const float* deb  = (const float*)d_in[13];

  float* out     = (float*)d_out;
  float* pooled  = out;                       // [64, 96]
  float* quant   = out + NB * 96;             // [64, 2003, 128]
  float* targets = quant + (long)M3 * 128;    // [64, 2003, 2]

  // ---- workspace layout (f16 activations, rows=(b,t), cols=channels) ----
  char* ws = (char*)d_ws;
  const size_t oH1 = 0;                            // M1 x 32
  const size_t oH2 = oH1 + (size_t)M1 * 32 * 2;    // M2 x 64
  const size_t oH3 = oH2 + (size_t)M2 * 64 * 2;    // M3 x 96
  const size_t oQ  = oH3 + (size_t)M3 * 96 * 2;    // M3 x 128
  const size_t oW  = oQ  + (size_t)M3 * 128 * 2;   // packed weights
  _Float16* h1 = (_Float16*)(ws + oH1);
  _Float16* h2 = (_Float16*)(ws + oH2);
  _Float16* h3 = (_Float16*)(ws + oH3);
  _Float16* q  = (_Float16*)(ws + oQ);
  // quantH overlays h1/h2 (both dead by the time wp_argmax runs)
  _Float16* quantH = (_Float16*)(ws + oH1);         // M3 x 128
  _Float16* w2p  = (_Float16*)(ws + oW);                      // 64  x 512
  _Float16* w3p  = (_Float16*)(ws + oW + 65536);              // 96  x 512
  _Float16* qewp = (_Float16*)(ws + oW + 65536 + 98304);      // 128 x 96
  _Float16* wpwp = (_Float16*)(ws + oW + 65536 + 98304 + 24576);           // 640 x 128
  _Float16* dewp = (_Float16*)(ws + oW + 65536 + 98304 + 24576 + 163840);  // 96 x 128

  init_pooled<<<(NB * 96 + 255) / 256, 256, 0, stream>>>(pooled);

  repack_conv_w<32, 16, 5><<<(64 * 512 + 255) / 256, 256, 0, stream>>>(c2w, w2p, 64);
  repack_conv_w<64, 8, 6><<<(96 * 512 + 255) / 256, 256, 0, stream>>>(c3w, w3p, 96);
  cvt_w_f16<<<(128 * 96 + 255) / 256, 256, 0, stream>>>(qew, qewp, 128 * 96);
  cvt_w_f16<<<(640 * 128 + 255) / 256, 256, 0, stream>>>(wpw, wpwp, 640 * 128);
  cvt_w_f16<<<(96 * 128 + 255) / 256, 256, 0, stream>>>(dew, dewp, 96 * 128);

  conv1_direct<<<(M1 + 255) / 256, 256, 0, stream>>>(x, c1w, c1b, h1);

  conv_gemm<32, 16, 4><<<M2 / 16, 32, 0, stream>>>(h1, w2p, c2b, h2, T1, T2, 64);
  conv_gemm<64, 8, 6><<<M3 / 16, 32, 0, stream>>>(h2, w3p, c3b, h3, T2, T3, 96);

  qe_gemm<<<M3 / 16, 32, 0, stream>>>(h3, qewp, qeb, q);

  wp_argmax_quant<<<M3 / 16, 32, 0, stream>>>(q, wpwp, wpb, cbk, quant, quantH, targets);

  de_pool<<<M3 / 16, 32, 0, stream>>>(quantH, dewp, deb, pooled);
}
